// FDATransformerEncoder_29128468201520
// MI455X (gfx1250) — matrix operand order repaired and checked by reading, not compiled, not run on hardware
//
#include <hip/hip_runtime.h>

typedef __bf16 bf16;
typedef __attribute__((ext_vector_type(16))) __bf16 v16bf;
typedef __attribute__((ext_vector_type(8)))  __bf16 bf16x8;
typedef __attribute__((ext_vector_type(8)))  float  v8f;

#define DEV __device__ __forceinline__

constexpr int BATCH = 8;
constexpr int NTOK  = 4096;             // 16*16*16 tokens per batch
constexpr int DIMM  = 512;
constexpr int INNER = 512;              // HEADS * DIM_HEAD
constexpr int HS    = 4;                // HEADS / 2
constexpr int DH    = 64;
constexpr int NTT   = 16;               // temporal length
constexpr int SSP   = 256;              // spatial length (NH*NW)
constexpr int MTOT  = BATCH * NTOK;     // 32768 rows
constexpr int QKVN  = 3 * INNER;        // 1536
constexpr int DEPTH = 4;

// ---------------- WMMA fragment helpers (CDNA5 16x16x32 bf16 layouts) ----------

// A-matrix 16x32 (MxK), row-major source with row = lane%16.
// lanes 0-15: K 0..7 & 16..23 ; lanes 16-31: K 8..15 & 24..31  -> two b128 loads.
DEV v16bf ldA16x32(const bf16* row, int hi) {
  bf16x8 a0 = *(const bf16x8*)(row + (hi ? 8 : 0));
  bf16x8 a1 = *(const bf16x8*)(row + 16 + (hi ? 8 : 0));
  v16bf r;
#pragma unroll
  for (int i = 0; i < 8; ++i) { r[i] = a0[i]; r[i + 8] = a1[i]; }
  return r;
}

// B-matrix 32x16 (KxN), from N-major (transposed) source: row = N = lane%16,
// lanes 0-15: K 0..7 & 16..23 ; lanes 16-31: K 8..15 & 24..31 -> two b128 loads.
DEV v16bf ldB32x16(const bf16* row, int hi) {
  bf16x8 a0 = *(const bf16x8*)(row + (hi ? 8 : 0));
  bf16x8 a1 = *(const bf16x8*)(row + 16 + (hi ? 8 : 0));
  v16bf r;
#pragma unroll
  for (int i = 0; i < 8; ++i) { r[i] = a0[i]; r[i + 8] = a1[i]; }
  return r;
}

DEV v8f wmma_bf16(v16bf a, v16bf b, v8f c) {
  return __builtin_amdgcn_wmma_f32_16x16x32_bf16(false, a, false, b, (short)0, c,
                                                 false, false);
}

// Async global->LDS 16-byte copy (per lane), tracked with ASYNCcnt.
// Flat shared-aperture addresses carry the LDS byte offset in the low 32 bits.
DEV void async_cp16(bf16* lds, const bf16* gptr) {
  unsigned l = (unsigned)(uintptr_t)lds;
  unsigned long long g = (unsigned long long)(uintptr_t)gptr;
  asm volatile("global_load_async_to_lds_b128 %0, %1, off"
               :: "v"(l), "v"(g) : "memory");
}
DEV void wait_async0() { asm volatile("s_wait_asynccnt 0x0" ::: "memory"); }

// ---------------- weight convert + transpose (f32 KxN -> bf16 NxK) -------------

__global__ void wconv_kernel(const float* __restrict__ W, bf16* __restrict__ Wt,
                             int rows /*K*/, int cols /*N*/) {
  int idx = blockIdx.x * 256 + threadIdx.x;
  if (idx < rows * cols) {
    int r = idx / cols, c = idx % cols;
    Wt[(size_t)c * rows + r] = (bf16)W[idx];
  }
}

// ---------------- LayerNorm (f32 in -> bf16 out), one wave per token ----------

__global__ __launch_bounds__(256) void ln_kernel(const float* __restrict__ x,
                                                 const float* __restrict__ g,
                                                 const float* __restrict__ bt,
                                                 bf16* __restrict__ y) {
  int tok  = blockIdx.x * 8 + threadIdx.y;
  int lane = threadIdx.x;
  const float* row = x + (size_t)tok * DIMM;
  float v[16];
  float s = 0.f;
#pragma unroll
  for (int i = 0; i < 16; ++i) { v[i] = row[lane + i * 32]; s += v[i]; }
#pragma unroll
  for (int m = 16; m >= 1; m >>= 1) s += __shfl_xor(s, m, 32);
  float mu  = s * (1.0f / 512.0f);
  float var = 0.f;
#pragma unroll
  for (int i = 0; i < 16; ++i) { float d = v[i] - mu; var += d * d; }
#pragma unroll
  for (int m = 16; m >= 1; m >>= 1) var += __shfl_xor(var, m, 32);
  var *= (1.0f / 512.0f);
  float inv = rsqrtf(var + 1e-5f);
  bf16* yr = y + (size_t)tok * DIMM;
#pragma unroll
  for (int i = 0; i < 16; ++i) {
    int c = lane + i * 32;
    yr[c] = (bf16)((v[i] - mu) * inv * g[c] + bt[c]);
  }
}

// ---------------- tiled WMMA GEMM: C[M,N] = A[M,512] * Bt[N,512]^T -------------
// Double-buffered async global->LDS staging overlapped with WMMA.
// MODE 0: store bf16 C.  MODE 1: C = acc + bias[col] + resid[row,col] -> f32.

template <int MODE>
__global__ __launch_bounds__(256) void gemm_kernel(
    const bf16* __restrict__ A, const bf16* __restrict__ Bt, int N,
    bf16* __restrict__ Cb, const float* __restrict__ bias,
    const float* __restrict__ resid, float* __restrict__ Cf) {
  constexpr int K = 512, BM = 128, BN = 128, BK = 32, LDT = BK + 8;  // 80B rows
  __shared__ alignas(16) bf16 As[2][BM * LDT];
  __shared__ alignas(16) bf16 Bs[2][BN * LDT];

  int tid = threadIdx.x, lane = tid & 31, wid = tid >> 5;
  int wm = wid >> 1, wn = wid & 1;     // 4x2 wave grid; wave tile 32x64
  int hi = (lane >= 16), lm = lane & 15;
  int m0 = blockIdx.y * BM, n0 = blockIdx.x * BN;

  v8f acc[2][4];
#pragma unroll
  for (int i = 0; i < 2; ++i)
#pragma unroll
    for (int j = 0; j < 4; ++j) acc[i][j] = (v8f){};

  int srow = tid >> 1;                 // 0..127
  int scol = (tid & 1) * 16;           // 0 or 16
  const bf16* gA = A + (size_t)(m0 + srow) * K + scol;
  const bf16* gB = Bt + (size_t)(n0 + srow) * K + scol;
  int lofs = srow * LDT + scol;

  // prologue: stage k-block 0 into buffer 0
  async_cp16(&As[0][lofs], gA);
  async_cp16(&As[0][lofs + 8], gA + 8);
  async_cp16(&Bs[0][lofs], gB);
  async_cp16(&Bs[0][lofs + 8], gB + 8);

  int buf = 0;
  for (int kb = 0; kb < K; kb += BK, buf ^= 1) {
    wait_async0();         // this wave's copies into `buf` have landed
    __syncthreads();       // everyone's copies landed; prior reads finished
    if (kb + BK < K) {     // overlap next stage's copies with this stage's math
      const bf16* nA = gA + kb + BK;
      const bf16* nB = gB + kb + BK;
      async_cp16(&As[buf ^ 1][lofs], nA);
      async_cp16(&As[buf ^ 1][lofs + 8], nA + 8);
      async_cp16(&Bs[buf ^ 1][lofs], nB);
      async_cp16(&Bs[buf ^ 1][lofs + 8], nB + 8);
    }

    v16bf af[2], bfv[4];
#pragma unroll
    for (int mt = 0; mt < 2; ++mt)
      af[mt] = ldA16x32(&As[buf][(wm * 32 + mt * 16 + lm) * LDT], hi);
#pragma unroll
    for (int nt = 0; nt < 4; ++nt)
      bfv[nt] = ldB32x16(&Bs[buf][(wn * 64 + nt * 16 + lm) * LDT], hi);
#pragma unroll
    for (int mt = 0; mt < 2; ++mt)
#pragma unroll
      for (int nt = 0; nt < 4; ++nt)
        acc[mt][nt] = wmma_bf16(af[mt], bfv[nt], acc[mt][nt]);
  }

#pragma unroll
  for (int mt = 0; mt < 2; ++mt)
#pragma unroll
    for (int nt = 0; nt < 4; ++nt)
#pragma unroll
      for (int r = 0; r < 8; ++r) {
        int row = m0 + wm * 32 + mt * 16 + r + (hi ? 8 : 0);
        int col = n0 + wn * 64 + nt * 16 + lm;
        float v = acc[mt][nt][r];
        if (MODE == 0) {
          Cb[(size_t)row * N + col] = (bf16)v;
        } else {
          v += bias[col] + resid[(size_t)row * N + col];
          Cf[(size_t)row * N + col] = v;
        }
      }
}

// ---------------- spatial attention: 256x256, dh=64, flash per 16-row tile ----
// grid = BATCH*HS*NT (512 blocks), 8 waves; wave handles Q tiles {wid, wid+8}.

__global__ __launch_bounds__(256) void attn_spatial_kernel(
    const bf16* __restrict__ qkv, bf16* __restrict__ out) {
  constexpr int VLD = SSP + 8;                    // 264 -> 528B rows
  __shared__ alignas(16) bf16 Vt[DH * VLD];       // V transposed (dh, key)
  __shared__ alignas(16) bf16 Pl[8 * 16 * 40];    // per-wave 16x32 P scratch

  int blk = blockIdx.x;
  int b   = blk / (HS * NTT);
  int r1  = blk % (HS * NTT);
  int h   = r1 / NTT;
  int t   = r1 % NTT;
  size_t tok0 = (size_t)b * NTOK + (size_t)t * SSP;
  const bf16* qbase = qkv + tok0 * QKVN + h * DH;
  const bf16* kbase = qbase + INNER;
  const bf16* vbase = qbase + 2 * INNER;

  int tid = threadIdx.x, lane = tid & 31, wid = tid >> 5;
  int hi = (lane >= 16), lm = lane & 15;

  {  // stage V^T: thread tid owns key row tid
    const bf16* vr = vbase + (size_t)tid * QKVN;
#pragma unroll 8
    for (int d = 0; d < DH; ++d) Vt[d * VLD + tid] = vr[d];
  }
  __syncthreads();

  const float scale = 0.125f;                     // DH^-0.5
  bf16* myP = &Pl[wid * 16 * 40];

  for (int qi = 0; qi < 2; ++qi) {
    int qt = wid + qi * 8;                        // Q tile 0..15
    const bf16* qrow = qbase + (size_t)(qt * 16 + lm) * QKVN;
    v16bf qa0 = ldA16x32(qrow, hi);               // dh 0..31
    v16bf qa1 = ldA16x32(qrow + 32, hi);          // dh 32..63

    v8f o[4];
#pragma unroll
    for (int dt = 0; dt < 4; ++dt) o[dt] = (v8f){};
    float rmax[8], rsum[8];
#pragma unroll
    for (int r = 0; r < 8; ++r) { rmax[r] = -1e30f; rsum[r] = 0.f; }

    for (int j = 0; j < SSP; j += 32) {
      const bf16* kr0 = kbase + (size_t)(j + lm) * QKVN;
      const bf16* kr1 = kbase + (size_t)(j + 16 + lm) * QKVN;
      v8f s0 = (v8f){}, s1 = (v8f){};
      s0 = wmma_bf16(qa0, ldB32x16(kr0, hi), s0);
      s0 = wmma_bf16(qa1, ldB32x16(kr0 + 32, hi), s0);
      s1 = wmma_bf16(qa0, ldB32x16(kr1, hi), s1);
      s1 = wmma_bf16(qa1, ldB32x16(kr1 + 32, hi), s1);

#pragma unroll
      for (int r = 0; r < 8; ++r) {
        float a = s0[r] * scale;
        float c = s1[r] * scale;
        float m = fmaxf(a, c);
#pragma unroll
        for (int sh = 8; sh >= 1; sh >>= 1) m = fmaxf(m, __shfl_xor(m, sh, 32));
        float nm    = fmaxf(rmax[r], m);
        float alpha = __expf(rmax[r] - nm);
        float e0    = __expf(a - nm);
        float e1    = __expf(c - nm);
        float ps    = e0 + e1;
#pragma unroll
        for (int sh = 8; sh >= 1; sh >>= 1) ps += __shfl_xor(ps, sh, 32);
        rsum[r] = rsum[r] * alpha + ps;
        rmax[r] = nm;
#pragma unroll
        for (int dt = 0; dt < 4; ++dt) o[dt][r] *= alpha;
        int prow = r + (hi ? 8 : 0);
        myP[prow * 40 + lm]      = (bf16)e0;
        myP[prow * 40 + 16 + lm] = (bf16)e1;
      }
      v16bf pf = ldA16x32(&myP[lm * 40], hi);
#pragma unroll
      for (int dt = 0; dt < 4; ++dt) {
        v16bf vf = ldB32x16(&Vt[(dt * 16 + lm) * VLD + j], hi);
        o[dt] = wmma_bf16(pf, vf, o[dt]);
      }
    }
#pragma unroll
    for (int dt = 0; dt < 4; ++dt)
#pragma unroll
      for (int r = 0; r < 8; ++r) {
        int row = qt * 16 + r + (hi ? 8 : 0);
        int col = h * DH + dt * 16 + lm;
        out[(tok0 + row) * INNER + col] = (bf16)(o[dt][r] / rsum[r]);
      }
  }
}

// ---------------- temporal attention: 16x16, dh=64, one wave per unit ---------
// grid = BATCH*HS*SSP/8 (1024 blocks); K padded 16->32 with zeros.

__global__ __launch_bounds__(256) void attn_temporal_kernel(
    const bf16* __restrict__ qkv, bf16* __restrict__ out) {
  constexpr int VLD2 = 40;
  __shared__ alignas(16) bf16 Vt[8 * DH * VLD2];  // per-wave (dh, key[32])
  __shared__ alignas(16) bf16 Pl[8 * 16 * 40];

  int tid = threadIdx.x, lane = tid & 31, wid = tid >> 5;
  int hi = (lane >= 16), lm = lane & 15;
  int unit = blockIdx.x * 8 + wid;
  int b    = unit / (HS * SSP);
  int r1   = unit % (HS * SSP);
  int h    = HS + r1 / SSP;                        // second half of heads
  int sidx = r1 % SSP;
  size_t tok0 = (size_t)b * NTOK + (size_t)sidx * NTT;
  const bf16* qbase = qkv + tok0 * QKVN + h * DH;
  const bf16* kbase = qbase + INNER;
  const bf16* vbase = qbase + 2 * INNER;
  bf16* myV = &Vt[wid * DH * VLD2];
  bf16* myP = &Pl[wid * 16 * 40];

  // stage V^T (16 valid keys, keys 16..31 zeroed)
  if (!hi) {
    const bf16* vr = vbase + (size_t)lm * QKVN;
#pragma unroll 8
    for (int d = 0; d < DH; ++d) myV[d * VLD2 + lm] = vr[d];
  } else {
#pragma unroll 8
    for (int d = 0; d < DH; ++d) myV[d * VLD2 + 16 + lm] = (bf16)0.f;
  }

  const bf16* qrow = qbase + (size_t)lm * QKVN;
  v16bf qa0 = ldA16x32(qrow, hi);
  v16bf qa1 = ldA16x32(qrow + 32, hi);
  const bf16* krow = kbase + (size_t)lm * QKVN;
  v8f s = (v8f){};
  s = wmma_bf16(qa0, ldB32x16(krow, hi), s);
  s = wmma_bf16(qa1, ldB32x16(krow + 32, hi), s);

  const float scale = 0.125f;
  float rsum[8];
#pragma unroll
  for (int r = 0; r < 8; ++r) {
    float a = s[r] * scale;
    float m = a;
#pragma unroll
    for (int sh = 8; sh >= 1; sh >>= 1) m = fmaxf(m, __shfl_xor(m, sh, 32));
    float e  = __expf(a - m);
    float ps = e;
#pragma unroll
    for (int sh = 8; sh >= 1; sh >>= 1) ps += __shfl_xor(ps, sh, 32);
    rsum[r] = ps;
    int prow = r + (hi ? 8 : 0);
    myP[prow * 40 + lm]      = (bf16)e;
    myP[prow * 40 + 16 + lm] = (bf16)0.f;        // pad K 16..31
  }

  v16bf pf = ldA16x32(&myP[lm * 40], hi);
#pragma unroll
  for (int dt = 0; dt < 4; ++dt) {
    v16bf vf = ldB32x16(&myV[(dt * 16 + lm) * VLD2], hi);
    v8f o = wmma_bf16(pf, vf, (v8f){});
#pragma unroll
    for (int r = 0; r < 8; ++r) {
      int row = r + (hi ? 8 : 0);
      int col = h * DH + dt * 16 + lm;
      out[(tok0 + row) * INNER + col] = (bf16)(o[r] / rsum[r]);
    }
  }
}

// ---------------- host-side orchestration -------------------------------------

extern "C" void kernel_launch(void* const* d_in, const int* in_sizes, int n_in,
                              void* d_out, int out_size, void* d_ws,
                              size_t ws_size, hipStream_t stream) {
  (void)in_sizes; (void)n_in; (void)out_size; (void)ws_size;
  const float* x_in  = (const float*)d_in[0];
  const float* Wqkv  = (const float*)d_in[1];
  const float* Wout  = (const float*)d_in[2];
  const float* bout  = (const float*)d_in[3];
  const float* gamma = (const float*)d_in[4];
  const float* beta  = (const float*)d_in[5];
  float* xout = (float*)d_out;

  char* ws = (char*)d_ws;
  size_t off = 0;
  float* xbuf   = (float*)(ws + off); off += (size_t)MTOT * DIMM * 4;
  bf16*  ybuf   = (bf16*)(ws + off);  off += (size_t)MTOT * DIMM * 2;
  bf16*  qkvbuf = (bf16*)(ws + off);  off += (size_t)MTOT * QKVN * 2;
  bf16*  abuf   = (bf16*)(ws + off);  off += (size_t)MTOT * INNER * 2;
  bf16*  wqkvT  = (bf16*)(ws + off);  off += (size_t)DIMM * QKVN * 2;
  bf16*  woutT  = (bf16*)(ws + off);  off += (size_t)INNER * DIMM * 2;

  const float* cur = x_in;
  for (int l = 0; l < DEPTH; ++l) {
    float* nxt = (l % 2 == 0) ? xbuf : xout;   // l=3 lands in d_out

    wconv_kernel<<<(DIMM * QKVN + 255) / 256, 256, 0, stream>>>(
        Wqkv + (size_t)l * DIMM * QKVN, wqkvT, DIMM, QKVN);
    wconv_kernel<<<(INNER * DIMM + 255) / 256, 256, 0, stream>>>(
        Wout + (size_t)l * INNER * DIMM, woutT, INNER, DIMM);

    ln_kernel<<<MTOT / 8, dim3(32, 8), 0, stream>>>(cur, gamma + l * DIMM,
                                                    beta + l * DIMM, ybuf);

    gemm_kernel<0><<<dim3(QKVN / 128, MTOT / 128), 256, 0, stream>>>(
        ybuf, wqkvT, QKVN, qkvbuf, nullptr, nullptr, nullptr);

    attn_spatial_kernel<<<BATCH * HS * NTT, 256, 0, stream>>>(qkvbuf, abuf);
    attn_temporal_kernel<<<BATCH * HS * SSP / 8, 256, 0, stream>>>(qkvbuf, abuf);

    gemm_kernel<1><<<dim3(DIMM / 128, MTOT / 128), 256, 0, stream>>>(
        abuf, woutT, DIMM, nullptr, bout + l * DIMM, cur, nxt);

    cur = nxt;
  }
}
